// ArcFace_30906584662273
// MI455X (gfx1250) — matrix operand order repaired and checked by reading, not compiled, not run on hardware
//
#include <hip/hip_runtime.h>
#include <hip/hip_bf16.h>

typedef __attribute__((ext_vector_type(16))) _Float16 v16h;
typedef __attribute__((ext_vector_type(8)))  _Float16 v8h;
typedef __attribute__((ext_vector_type(8)))  float    v8f;

// ---------------- problem constants ----------------
#define NB      512
#define NCLS    100000
#define DFEAT   128
// conv1: 1->16, 4x4 s2 p1, 112 -> 56
#define C1 16
#define H1 56
// conv2: 16->32, 4x4 s2 p1, 56 -> 28
#define C2 32
#define H2 28
// conv3: 32->64, 3x3 s1 p0, 28 -> 26
#define C3 64
#define H3 26
// conv4: 64->128, 3x3 s1 p0, 26 -> 24
#define C4 128
#define H4 24

#define S1 ((size_t)NB * C1 * H1 * H1)   // 25,690,112 floats
#define S2 ((size_t)NB * C2 * H2 * H2)   // 12,845,056
#define S3 ((size_t)NB * C3 * H3 * H3)   // 22,151,168
#define S4 ((size_t)NB * C4 * H4 * H4)   // 37,748,736

// conv4 implicit GEMM: M x N x K
#define GM4 ((size_t)NB * H4 * H4)       // 294912 (divisible by 128)
#define GK4 (C3 * 9)                     // 576   (divisible by 32)
#define AS_STRIDE 40                     // padded A-slab row stride (halves)

// ---------------- conv1: direct fp32 ----------------
__global__ void conv1_kernel(const float* __restrict__ x, const float* __restrict__ w,
                             float* __restrict__ y) {
  size_t o = (size_t)blockIdx.x * 256 + threadIdx.x;
  if (o >= S1) return;
  int ox = (int)(o % H1); size_t t = o / H1;
  int oy = (int)(t % H1); t /= H1;
  int c  = (int)(t % C1); int img = (int)(t / C1);
  const float* xi = x + (size_t)img * 112 * 112;
  const float* wf = w + c * 16;                 // (16,1,4,4)
  float s = 0.f;
#pragma unroll
  for (int ky = 0; ky < 4; ++ky) {
    int iy = oy * 2 - 1 + ky;
    if (iy < 0 || iy >= 112) continue;
#pragma unroll
    for (int kx = 0; kx < 4; ++kx) {
      int ix = ox * 2 - 1 + kx;
      if (ix < 0 || ix >= 112) continue;
      s += xi[iy * 112 + ix] * wf[ky * 4 + kx];
    }
  }
  y[o] = s;
}

// ---------------- conv2: direct fp32 ----------------
__global__ void conv2_kernel(const float* __restrict__ xin, const float* __restrict__ w,
                             float* __restrict__ y) {
  size_t o = (size_t)blockIdx.x * 256 + threadIdx.x;
  if (o >= S2) return;
  int ox = (int)(o % H2); size_t t = o / H2;
  int oy = (int)(t % H2); t /= H2;
  int oc = (int)(t % C2); int img = (int)(t / C2);
  const float* xi = xin + (size_t)img * C1 * H1 * H1;
  const float* wf = w + (size_t)oc * C1 * 16;   // (32,16,4,4)
  float s = 0.f;
  for (int ci = 0; ci < C1; ++ci) {
#pragma unroll
    for (int ky = 0; ky < 4; ++ky) {
      int iy = oy * 2 - 1 + ky;
      if (iy < 0 || iy >= H1) continue;
#pragma unroll
      for (int kx = 0; kx < 4; ++kx) {
        int ix = ox * 2 - 1 + kx;
        if (ix < 0 || ix >= H1) continue;
        s += xi[(ci * H1 + iy) * H1 + ix] * wf[ci * 16 + ky * 4 + kx];
      }
    }
  }
  y[o] = s;
}

// ---------------- conv3: direct fp32 (no pad) ----------------
__global__ void conv3_kernel(const float* __restrict__ xin, const float* __restrict__ w,
                             float* __restrict__ y) {
  size_t o = (size_t)blockIdx.x * 256 + threadIdx.x;
  if (o >= S3) return;
  int ox = (int)(o % H3); size_t t = o / H3;
  int oy = (int)(t % H3); t /= H3;
  int oc = (int)(t % C3); int img = (int)(t / C3);
  const float* xi = xin + (size_t)img * C2 * H2 * H2 + oy * H2 + ox;
  const float* wf = w + (size_t)oc * C2 * 9;    // (64,32,3,3)
  float s = 0.f;
  for (int ci = 0; ci < C2; ++ci) {
#pragma unroll
    for (int ky = 0; ky < 3; ++ky)
#pragma unroll
      for (int kx = 0; kx < 3; ++kx)
        s += xi[(ci * H2 + ky) * H2 + kx] * wf[ci * 9 + ky * 3 + kx];
  }
  y[o] = s;
}

// ---------------- conv4: implicit-GEMM WMMA f16 (f32 accum) ----------------
// M = 294912 (img,oy,ox), N = 128 (out channels), K = 576 (ci,ky,kx).
// Block = 256 threads = 8 waves; block tile = 128(M) x 16(N).
// Per K-step the im2col A slab (128x32) is cooperatively staged in LDS with
// coalesced loads; the k->spatial offset table removes all div/mod chains.
__global__ void conv4_wmma_kernel(const float* __restrict__ y3,
                                  const float* __restrict__ w4,
                                  float* __restrict__ y4) {
  __shared__ __attribute__((aligned(32))) _Float16 Bs[16 * GK4];        // 18 KB [n][k]
  __shared__ __attribute__((aligned(32))) _Float16 As[128 * AS_STRIDE]; // 10 KB [m][kk]
  __shared__ int Dtab[GK4];                                             // 2.25 KB
  const int tid = threadIdx.x;
  const int n0  = blockIdx.y * 16;

  for (int idx = tid; idx < 16 * GK4; idx += 256) {
    int n = idx / GK4, k = idx - n * GK4;       // k = ci*9 + ky*3 + kx
    Bs[idx] = (_Float16)w4[(size_t)(n0 + n) * GK4 + k];
  }
  for (int k = tid; k < GK4; k += 256) {
    int ci = k / 9, r = k - ci * 9;
    int ky = r / 3, kx = r - (r / 3) * 3;
    Dtab[k] = (ci * H3 + ky) * H3 + kx;         // im2col input offset for k
  }

  // Staging constants: this thread always loads row (tid&127) of the M tile,
  // at K columns kk = (tid>>7) + 2*i, i = 0..15.
  const int mloc = tid & 127;
  const int kb   = tid >> 7;
  int off_m;
  {
    int mrow = blockIdx.x * 128 + mloc;
    int img = mrow / (H4 * H4);
    int pos = mrow - img * (H4 * H4);
    int oy = pos / H4, ox = pos - (pos / H4) * H4;
    off_m = (img * C3 * H3 + oy) * H3 + ox;     // fits in int (max ~22.1M)
  }

  // Fragment constants (A: lane L holds M = L%16; B: lane holds N = L%16)
  const int lane = tid & 31;
  const int wave = tid >> 5;
  const int arow = (wave * 16 + (lane & 15)) * AS_STRIDE;
  const int khiA = (lane >= 16) ? 16 : 0;
  const _Float16* brow = Bs + (size_t)(lane & 15) * GK4 + ((lane >= 16) ? 16 : 0);

  __syncthreads();                              // Bs, Dtab ready

  v8f acc = {};
  for (int k0 = 0; k0 < GK4; k0 += 32) {
    // stage A slab: coalesced (consecutive threads -> consecutive spatial m)
#pragma unroll
    for (int i = 0; i < 16; ++i) {
      int kk = kb + 2 * i;
      As[mloc * AS_STRIDE + kk] = (_Float16)y3[off_m + Dtab[k0 + kk]];
    }
    if (k0 + 32 < GK4)                          // overlap next slab's HBM latency
      __builtin_prefetch((const void*)(y3 + off_m + Dtab[k0 + 32 + kb]), 0, 1);
    __syncthreads();

    v8h alo = *(const v8h*)(As + arow + khiA);       // conflict-free ds_load_b128
    v8h ahi = *(const v8h*)(As + arow + khiA + 8);
    v16h a;
#pragma unroll
    for (int j = 0; j < 8; ++j) { a[j] = alo[j]; a[8 + j] = ahi[j]; }
    v16h b = *(const v16h*)(brow + k0);
    acc = __builtin_amdgcn_wmma_f32_16x16x32_f16(false, a, false, b,
                                                 (short)0, acc, false, false);
    __syncthreads();                            // protect As before restaging
  }

  const int mbase = blockIdx.x * 128 + wave * 16;
  const int ncol  = n0 + (lane & 15);
#pragma unroll
  for (int v = 0; v < 8; ++v) {
    int row = mbase + v + ((lane >= 16) ? 8 : 0);
    int im = row / (H4 * H4);
    int p  = row - im * (H4 * H4);
    int ry = p / H4, rx = p - (p / H4) * H4;
    y4[(((size_t)im * C4 + ncol) * H4 + ry) * H4 + rx] = acc[v];
  }
}

// ---------------- BN batch stats (training mode, biased var) ----------------
__global__ void bn_stats_kernel(const float* __restrict__ y, float* __restrict__ mu,
                                float* __restrict__ rstd, int C, int HW) {
  __shared__ float ssum[256], ssq[256];
  const int c = blockIdx.x;
  const size_t CHW = (size_t)C * HW;
  float s = 0.f, q = 0.f;
  for (int img = 0; img < NB; ++img) {
    const float* p = y + (size_t)img * CHW + (size_t)c * HW;
    for (int i = threadIdx.x; i < HW; i += 256) { float v = p[i]; s += v; q += v * v; }
  }
  ssum[threadIdx.x] = s; ssq[threadIdx.x] = q;
  __syncthreads();
  for (int off = 128; off > 0; off >>= 1) {
    if (threadIdx.x < off) {
      ssum[threadIdx.x] += ssum[threadIdx.x + off];
      ssq[threadIdx.x]  += ssq[threadIdx.x + off];
    }
    __syncthreads();
  }
  if (threadIdx.x == 0) {
    float inv = 1.0f / ((float)NB * (float)HW);
    float m = ssum[0] * inv;
    float var = ssq[0] * inv - m * m;
    mu[c] = m;
    rstd[c] = rsqrtf(var + 1e-5f);
  }
}

__global__ void bn_relu_kernel(float* __restrict__ y, const float* __restrict__ mu,
                               const float* __restrict__ rstd,
                               const float* __restrict__ g, const float* __restrict__ be,
                               int C, int HW, size_t total) {
  size_t o = (size_t)blockIdx.x * 256 + threadIdx.x;
  if (o >= total) return;
  int c = (int)((o / HW) % C);
  float v = (y[o] - mu[c]) * rstd[c] * g[c] + be[c];
  y[o] = v > 0.f ? v : 0.f;
}

// ---------------- global average pool ----------------
__global__ void gap_kernel(const float* __restrict__ y4, float* __restrict__ f) {
  int o = blockIdx.x * 256 + threadIdx.x;       // o = img*128 + c
  if (o >= NB * DFEAT) return;
  const float* p = y4 + (size_t)o * (H4 * H4);
  float s = 0.f;
  for (int i = 0; i < H4 * H4; ++i) s += p[i];
  f[o] = s * (1.0f / (H4 * H4));
}

// ---------------- L2 row-normalize (128-wide) to f16 ----------------
__global__ void l2norm_f16_kernel(const float* __restrict__ src, _Float16* __restrict__ dst) {
  __shared__ float red[128];
  const int r = blockIdx.x;
  float v = src[(size_t)r * DFEAT + threadIdx.x];
  red[threadIdx.x] = v * v;
  __syncthreads();
  for (int off = 64; off > 0; off >>= 1) {
    if ((int)threadIdx.x < off) red[threadIdx.x] += red[threadIdx.x + off];
    __syncthreads();
  }
  float nrm = fmaxf(sqrtf(red[0]), 1e-12f);
  dst[(size_t)r * DFEAT + threadIdx.x] = (_Float16)(v / nrm);
}

// ---------------- ArcFace GEMM: 64 * (fn @ wn^T), WMMA f16 ----------------
// M=512, N=100000 (6250 tiles of 16), K=128. Block = 8 waves; each wave owns
// one 16x16 output tile (wave-uniform tail guard keeps EXEC all-ones).
__global__ void arc_gemm_kernel(const _Float16* __restrict__ fn,  // 512x128
                                const _Float16* __restrict__ wn,  // 100000x128
                                float* __restrict__ out) {
  const int lane = threadIdx.x & 31;
  const int wave = threadIdx.x >> 5;
  const int ntile = blockIdx.x * 8 + wave;
  if (ntile >= NCLS / 16) return;               // wave-uniform
  const int n0 = ntile * 16;
  const int m0 = blockIdx.y * 16;
  const int mrow = m0 + (lane & 15);
  const int khiA = (lane >= 16) ? 16 : 0;
  const int khiB = (lane >= 16) ? 16 : 0;
  const int ncol = n0 + (lane & 15);
  const _Float16* arow = fn + (size_t)mrow * DFEAT;
  const _Float16* brow = wn + (size_t)ncol * DFEAT + khiB;
  v8f acc = {};
#pragma unroll
  for (int k0 = 0; k0 < DFEAT; k0 += 32) {
    v8h lo = *(const v8h*)(arow + k0 + khiA);
    v8h hi = *(const v8h*)(arow + k0 + khiA + 8);
    v16h a;
#pragma unroll
    for (int j = 0; j < 8; ++j) { a[j] = lo[j]; a[8 + j] = hi[j]; }
    v16h b = *(const v16h*)(brow + k0);         // contiguous 32B global read
    acc = __builtin_amdgcn_wmma_f32_16x16x32_f16(false, a, false, b,
                                                 (short)0, acc, false, false);
  }
#pragma unroll
  for (int v = 0; v < 8; ++v) {
    int row = m0 + v + ((lane >= 16) ? 8 : 0);
    out[(size_t)row * NCLS + ncol] = 64.0f * acc[v];
  }
}

// ---------------- ArcFace margin scatter on target class ----------------
__global__ void margin_kernel(const int* __restrict__ gt, float* __restrict__ out) {
  int r = blockIdx.x * 256 + threadIdx.x;
  if (r >= NB) return;
  size_t idx = (size_t)r * NCLS + gt[r];
  float c = out[idx] * (1.0f / 64.0f);
  c = fminf(fmaxf(c, -1.0f), 1.0f);
  out[idx] = 64.0f * cosf(acosf(c) + 0.5f);
}

// ---------------- launcher ----------------
extern "C" void kernel_launch(void* const* d_in, const int* in_sizes, int n_in,
                              void* d_out, int out_size, void* d_ws, size_t ws_size,
                              hipStream_t stream) {
  (void)in_sizes; (void)n_in; (void)out_size; (void)ws_size;
  const float* x     = (const float*)d_in[0];
  const int*   gt    = (const int*)  d_in[1];
  const float* w1    = (const float*)d_in[2];
  const float* g1    = (const float*)d_in[4];
  const float* be1   = (const float*)d_in[5];
  const float* w2    = (const float*)d_in[6];
  const float* g2    = (const float*)d_in[8];
  const float* be2   = (const float*)d_in[9];
  const float* w3    = (const float*)d_in[10];
  const float* g3    = (const float*)d_in[12];
  const float* be3   = (const float*)d_in[13];
  const float* w4    = (const float*)d_in[14];
  const float* g4    = (const float*)d_in[16];
  const float* be4   = (const float*)d_in[17];
  const float* arc_w = (const float*)d_in[18];
  float* out = (float*)d_out;
  // (conv biases b1..b4 cancel under train-mode BatchNorm: mu absorbs them)

  char* ws = (char*)d_ws;
  float* y1 = (float*)(ws);                                   // S1 floats @ 0
  float* y2 = (float*)(ws + S1 * 4);                          // S2 floats
  float* y3 = (float*)(ws);                                   // reuses y1 (y1 dead)
  float* y4 = (float*)(ws + (S1 + S2) * 4);                   // S4 floats
  size_t off = (S1 + S2 + S4) * 4;
  float*     f     = (float*)(ws + off);      off += (size_t)NB * DFEAT * 4;
  _Float16*  fnorm = (_Float16*)(ws + off);   off += (size_t)NB * DFEAT * 2;
  _Float16*  wnorm = (_Float16*)(ws + off);   off += (size_t)NCLS * DFEAT * 2;
  float*     mu    = (float*)(ws + off);      off += 512;
  float*     rstd  = (float*)(ws + off);

  dim3 blk(256);

  // stage 1
  conv1_kernel<<<dim3((unsigned)((S1 + 255) / 256)), blk, 0, stream>>>(x, w1, y1);
  bn_stats_kernel<<<dim3(C1), blk, 0, stream>>>(y1, mu, rstd, C1, H1 * H1);
  bn_relu_kernel<<<dim3((unsigned)((S1 + 255) / 256)), blk, 0, stream>>>(y1, mu, rstd, g1, be1, C1, H1 * H1, S1);
  // stage 2
  conv2_kernel<<<dim3((unsigned)((S2 + 255) / 256)), blk, 0, stream>>>(y1, w2, y2);
  bn_stats_kernel<<<dim3(C2), blk, 0, stream>>>(y2, mu, rstd, C2, H2 * H2);
  bn_relu_kernel<<<dim3((unsigned)((S2 + 255) / 256)), blk, 0, stream>>>(y2, mu, rstd, g2, be2, C2, H2 * H2, S2);
  // stage 3 (y3 overwrites y1's region; y1 is dead)
  conv3_kernel<<<dim3((unsigned)((S3 + 255) / 256)), blk, 0, stream>>>(y2, w3, y3);
  bn_stats_kernel<<<dim3(C3), blk, 0, stream>>>(y3, mu, rstd, C3, H3 * H3);
  bn_relu_kernel<<<dim3((unsigned)((S3 + 255) / 256)), blk, 0, stream>>>(y3, mu, rstd, g3, be3, C3, H3 * H3, S3);
  // stage 4: WMMA implicit GEMM
  conv4_wmma_kernel<<<dim3((unsigned)(GM4 / 128), C4 / 16), blk, 0, stream>>>(y3, w4, y4);
  bn_stats_kernel<<<dim3(C4), blk, 0, stream>>>(y4, mu, rstd, C4, H4 * H4);
  bn_relu_kernel<<<dim3((unsigned)((S4 + 255) / 256)), blk, 0, stream>>>(y4, mu, rstd, g4, be4, C4, H4 * H4, S4);
  // head
  gap_kernel<<<dim3((NB * DFEAT + 255) / 256), blk, 0, stream>>>(y4, f);
  l2norm_f16_kernel<<<dim3(NB), dim3(128), 0, stream>>>(f, fnorm);
  l2norm_f16_kernel<<<dim3(NCLS), dim3(128), 0, stream>>>(arc_w, wnorm);
  arc_gemm_kernel<<<dim3((NCLS / 16 + 7) / 8, NB / 16), blk, 0, stream>>>(fnorm, wnorm, out);
  margin_kernel<<<dim3((NB + 255) / 256), blk, 0, stream>>>(gt, out);
}
